// GCNNet_84086869721772
// MI455X (gfx1250) — matrix operand order repaired and checked by reading, not compiled, not run on hardware
//
#include <hip/hip_runtime.h>
#include <hip/hip_bf16.h>

// ---------------- WMMA types ----------------
typedef __attribute__((ext_vector_type(16))) __bf16 v16bf;
typedef __attribute__((ext_vector_type(8)))  __bf16 v8bf;
typedef __attribute__((ext_vector_type(8)))  float  v8f;

#define BM 64
#define BN 64
#define BK 32
#define LDSK 40           // padded row stride (halves): 80B -> every 8-half run 16B aligned
#define GEMM_THREADS 128

static inline int ceil_div(int a, int b) { return (a + b - 1) / b; }

__device__ __forceinline__ v16bf concat8(v8bf a, v8bf b) {
    return __builtin_shufflevector(a, b, 0, 1, 2, 3, 4, 5, 6, 7, 8, 9, 10, 11, 12, 13, 14, 15);
}

union pack2bf { unsigned u; __bf16 h[2]; };

// split one f32 pair into packed hi / lo bf16 words
__device__ __forceinline__ void split2(float v0, float v1, unsigned& hi, unsigned& lo) {
    pack2bf ph, pl;
    ph.h[0] = (__bf16)v0;
    ph.h[1] = (__bf16)v1;
    pl.h[0] = (__bf16)(v0 - (float)ph.h[0]);
    pl.h[1] = (__bf16)(v1 - (float)ph.h[1]);
    hi = ph.u; lo = pl.u;
}

// ---------------- WMMA GEMM: C[M,ldc] = act(A[M,K] @ B[K,N] + bias) ----------------
// bf16 split (hi/lo) x 3 WMMAs per tile for ~fp32 accuracy on the bf16 16x16x32 pipe.
// NOTE: assumes K is even (true for every GEMM in this network) for 8B-aligned paired loads.
__global__ __launch_bounds__(GEMM_THREADS)
void gemm_wmma_kernel(const float* __restrict__ A, const float* __restrict__ B,
                      const float* __restrict__ bias, float* __restrict__ C,
                      int M, int Ncols, int K, int ldc, int act)
{
    // A staged [m][k]; B staged TRANSPOSED [n][k] so fragments are contiguous runs.
    __shared__ __align__(16) __bf16 sAh[BM][LDSK];
    __shared__ __align__(16) __bf16 sAl[BM][LDSK];
    __shared__ __align__(16) __bf16 sBh[BN][LDSK];
    __shared__ __align__(16) __bf16 sBl[BN][LDSK];

    const int tid  = threadIdx.x;
    const int lane = tid & 31;
    const int wave = tid >> 5;
    const int wm   = (wave & 1) * 32;   // wave's 32x32 quadrant
    const int wn   = (wave >> 1) * 32;
    const int m0   = blockIdx.y * BM;
    const int n0   = blockIdx.x * BN;
    const int l15  = lane & 15;
    const int g    = lane >> 4;

    const bool fullM = (m0 + BM <= M);
    const bool fullN = (n0 + BN <= Ncols);

    v8f acc[2][2] = {};

    for (int k0 = 0; k0 < K; k0 += BK) {
        const bool fullK = (k0 + BK <= K);

        // ---- stage A tile 64x32 (f32 -> bf16 hi/lo) ----
        if (fullM && fullK) {
            // branch-free fast path: 8 paired loads per thread, packed 32-bit LDS stores
#pragma unroll
            for (int i = 0; i < (BM * BK / 2) / GEMM_THREADS; ++i) {
                int idx = tid + i * GEMM_THREADS;       // 1024 pairs
                int m = idx >> 4;                       // 16 pairs per row
                int kp = (idx & 15) * 2;
                float2 v = *(const float2*)&A[(size_t)(m0 + m) * K + k0 + kp];
                unsigned hi, lo;
                split2(v.x, v.y, hi, lo);
                *(unsigned*)&sAh[m][kp] = hi;
                *(unsigned*)&sAl[m][kp] = lo;
            }
        } else {
#pragma unroll
            for (int i = 0; i < (BM * BK) / GEMM_THREADS; ++i) {
                int idx = tid + i * GEMM_THREADS;
                int m = idx >> 5, k = idx & 31;
                int gr = m0 + m, gc = k0 + k;
                float v = (gr < M && gc < K) ? A[(size_t)gr * K + gc] : 0.0f;
                __bf16 hi = (__bf16)v;
                __bf16 lo = (__bf16)(v - (float)hi);
                sAh[m][k] = hi; sAl[m][k] = lo;
            }
        }

        // ---- stage B tile 32x64 -> transposed LDS [n][k] ----
        if (fullN && fullK) {
#pragma unroll
            for (int i = 0; i < (BK / 2 * BN) / GEMM_THREADS; ++i) {
                int idx = tid + i * GEMM_THREADS;       // 1024 pairs
                int n = idx & 63;                       // consecutive tid -> consecutive n (coalesced)
                int kp = (idx >> 6) * 2;
                const float* bp = &B[(size_t)(k0 + kp) * Ncols + n0 + n];
                float v0 = bp[0];
                float v1 = bp[Ncols];
                unsigned hi, lo;
                split2(v0, v1, hi, lo);
                *(unsigned*)&sBh[n][kp] = hi;
                *(unsigned*)&sBl[n][kp] = lo;
            }
        } else {
#pragma unroll
            for (int i = 0; i < (BK * BN) / GEMM_THREADS; ++i) {
                int idx = tid + i * GEMM_THREADS;
                int k = idx >> 6, n = idx & 63;
                int gr = k0 + k, gc = n0 + n;
                float v = (gr < K && gc < Ncols) ? B[(size_t)gr * Ncols + gc] : 0.0f;
                __bf16 hi = (__bf16)v;
                __bf16 lo = (__bf16)(v - (float)hi);
                sBh[n][k] = hi; sBl[n][k] = lo;
            }
        }

        // prefetch next A tile rows (emits global_prefetch_b8)
        if (k0 + BK < K) {
            int pr = m0 + (tid >> 1);
            if (pr < M) __builtin_prefetch(&A[(size_t)pr * K + k0 + BK], 0, 0);
        }
        __syncthreads();

        // ---- B fragments for both tn, loaded once per K-step ----
        // B k order used here (not AMD's table order): lane l: n = l&15 ; K = 16*(l>>4) + e  -> contiguous 16-half run
        v16bf bh[2], bl[2];
#pragma unroll
        for (int tn = 0; tn < 2; ++tn) {
            int n = wn + tn * 16 + l15;
            const v8bf* pbh = (const v8bf*)(&sBh[n][0]);
            const v8bf* pbl = (const v8bf*)(&sBl[n][0]);
            bh[tn] = concat8(pbh[2 * g], pbh[2 * g + 1]);
            bl[tn] = concat8(pbl[2 * g], pbl[2 * g + 1]);
        }

#pragma unroll
        for (int tm = 0; tm < 2; ++tm) {
            // A k order used here, same as B: lane l: m = l&15 ; g = l>>4 ;
            // K = 16g + e -> contiguous 16-half run (matches B)
            int m = wm + tm * 16 + l15;
            const v8bf* pah = (const v8bf*)(&sAh[m][0]);
            const v8bf* pal = (const v8bf*)(&sAl[m][0]);
            v16bf ah = concat8(pah[2 * g], pah[2 * g + 1]);
            v16bf al = concat8(pal[2 * g], pal[2 * g + 1]);
#pragma unroll
            for (int tn = 0; tn < 2; ++tn) {
                acc[tm][tn] = __builtin_amdgcn_wmma_f32_16x16x32_bf16(
                    false, ah, false, bh[tn], (short)0, acc[tm][tn], false, false);
                acc[tm][tn] = __builtin_amdgcn_wmma_f32_16x16x32_bf16(
                    false, ah, false, bl[tn], (short)0, acc[tm][tn], false, false);
                acc[tm][tn] = __builtin_amdgcn_wmma_f32_16x16x32_bf16(
                    false, al, false, bh[tn], (short)0, acc[tm][tn], false, false);
            }
        }
        __syncthreads();
    }

    // ---- store: C/D 16x16 layout: lane l col n=l&15 ; VGPR r row = r + 8*(l>>4) ----
#pragma unroll
    for (int tm = 0; tm < 2; ++tm)
#pragma unroll
        for (int tn = 0; tn < 2; ++tn) {
            int n = n0 + wn + tn * 16 + l15;
            if (n >= Ncols) continue;
            float bv = bias ? bias[n] : 0.0f;
#pragma unroll
            for (int r = 0; r < 8; ++r) {
                int m = m0 + wm + tm * 16 + r + 8 * g;
                if (m < M) {
                    float v = acc[tm][tn][r] + bv;
                    if (act) v = fmaxf(v, 0.0f);
                    C[(size_t)m * ldc + n] = v;
                }
            }
        }
}

// ---------------- elementwise / graph kernels ----------------
__global__ void fill_kernel(float* __restrict__ p, float v, int n) {
    int i = blockIdx.x * blockDim.x + threadIdx.x;
    if (i < n) p[i] = v;
}

__global__ void deg_count_kernel(const int* __restrict__ dst, float* __restrict__ deg, int E) {
    int i = blockIdx.x * blockDim.x + threadIdx.x;
    if (i < E) atomicAdd(&deg[dst[i]], 1.0f);
}

__global__ void rsqrt_kernel(const float* __restrict__ deg, float* __restrict__ dinv, int n) {
    int i = blockIdx.x * blockDim.x + threadIdx.x;
    if (i < n) dinv[i] = rsqrtf(deg[i]);
}

// out[i,f] = h[i,f] * dinv[i]^2   (self-loop contribution; also initializes out). 2D grid: y=row.
__global__ void self_init_kernel(const float* __restrict__ h, const float* __restrict__ dinv,
                                 float* __restrict__ out, int F) {
    int i = blockIdx.y;
    int f = blockIdx.x * blockDim.x + threadIdx.x;
    if (f < F) {
        float s = dinv[i];
        out[(size_t)i * F + f] = h[(size_t)i * F + f] * s * s;
    }
}

// one wave32 per edge; lanes stride features; f32 atomics resolve in L2
__global__ void edge_scatter_kernel(const float* __restrict__ h, const float* __restrict__ dinv,
                                    const int* __restrict__ src, const int* __restrict__ dst,
                                    float* __restrict__ out, int E, int F) {
    int gtid = blockIdx.x * blockDim.x + threadIdx.x;
    int e = gtid >> 5;
    int lane = gtid & 31;
    if (e >= E) return;
    int s = src[e], d = dst[e];
    float norm = dinv[s] * dinv[d];
    const float* hs = h + (size_t)s * F;
    float* od = out + (size_t)d * F;
    for (int f = lane; f < F; f += 32) atomicAdd(od + f, hs[f] * norm);
}

// x[i,f] = relu(x[i,f] + b[f]). 2D grid: y=row.
__global__ void bias_relu_kernel(float* __restrict__ x, const float* __restrict__ b, int F) {
    int i = blockIdx.y;
    int f = blockIdx.x * blockDim.x + threadIdx.x;
    if (f < F) {
        size_t idx = (size_t)i * F + f;
        x[idx] = fmaxf(x[idx] + b[f], 0.0f);
    }
}

// segment max of non-negative (post-ReLU) values via int-punned atomicMax; pool pre-zeroed. 2D grid.
__global__ void seg_max_kernel(const float* __restrict__ x, const int* __restrict__ batch,
                               float* __restrict__ pool, int F) {
    int i = blockIdx.y;
    int f = blockIdx.x * blockDim.x + threadIdx.x;
    if (f < F) {
        int gseg = batch[i];
        atomicMax((int*)&pool[(size_t)gseg * F + f], __float_as_int(x[(size_t)i * F + f]));
    }
}

// row-wise L2 normalize: out[r,:] = cell[r,:] / max(||cell[r,:]||, 1e-12)
__global__ void l2norm_kernel(const float* __restrict__ cell, float* __restrict__ out, int cols) {
    __shared__ float red[256];
    int r = blockIdx.x;
    float s = 0.0f;
    for (int c = threadIdx.x; c < cols; c += blockDim.x) {
        float v = cell[(size_t)r * cols + c];
        s += v * v;
    }
    red[threadIdx.x] = s;
    __syncthreads();
    for (int off = 128; off > 0; off >>= 1) {
        if (threadIdx.x < off) red[threadIdx.x] += red[threadIdx.x + off];
        __syncthreads();
    }
    float inv = 1.0f / fmaxf(sqrtf(red[0]), 1e-12f);
    for (int c = threadIdx.x; c < cols; c += blockDim.x)
        out[(size_t)r * cols + c] = cell[(size_t)r * cols + c] * inv;
}

// ---------------- host-side orchestration ----------------
static void run_gemm(const float* A, const float* B, const float* bias, float* C,
                     int M, int N, int K, int ldc, int act, hipStream_t s) {
    dim3 grid(ceil_div(N, BN), ceil_div(M, BM));
    gemm_wmma_kernel<<<grid, GEMM_THREADS, 0, s>>>(A, B, bias, C, M, N, K, ldc, act);
}

static void drug_branch(const float* x, const int* ei, const int* batch,
                        const float* Wc1, const float* bc1, const float* Wc2, const float* bc2,
                        const float* Wc3, const float* bc3,
                        const float* Wg1, const float* bg1, const float* Wg2, const float* bg2,
                        float* t0, float* t1, float* deg, float* dinv, float* pool, float* gg,
                        float* xc_out, int N, int E, int G, hipStream_t s) {
    const int* src = ei;
    const int* dst = ei + E;

    // deg = in-degree + 1 (self loops); dinv = deg^{-1/2}
    fill_kernel<<<ceil_div(N, 256), 256, 0, s>>>(deg, 1.0f, N);
    deg_count_kernel<<<ceil_div(E, 256), 256, 0, s>>>(dst, deg, E);
    rsqrt_kernel<<<ceil_div(N, 256), 256, 0, s>>>(deg, dinv, N);

    struct { const float* W; const float* b; int Kin; int Fout; } L[3] = {
        { Wc1, bc1, 78, 78 }, { Wc2, bc2, 78, 156 }, { Wc3, bc3, 156, 312 } };

    const float* cur = x;
    for (int li = 0; li < 3; ++li) {
        int Kin = L[li].Kin, F = L[li].Fout;
        run_gemm(cur, L[li].W, nullptr, t0, N, F, Kin, F, 0, s);
        dim3 g2(ceil_div(F, 128), N);
        self_init_kernel<<<g2, 128, 0, s>>>(t0, dinv, t1, F);
        edge_scatter_kernel<<<ceil_div(E * 32, 256), 256, 0, s>>>(t0, dinv, src, dst, t1, E, F);
        bias_relu_kernel<<<g2, 128, 0, s>>>(t1, L[li].b, F);
        cur = t1;
    }

    // global max pool -> [G, 312]
    fill_kernel<<<ceil_div(G * 312, 256), 256, 0, s>>>(pool, 0.0f, G * 312);
    dim3 gp(ceil_div(312, 128), N);
    seg_max_kernel<<<gp, 128, 0, s>>>(t1, batch, pool, 312);

    // graph head: 312 -> 156 (relu) -> 128, writing into concat buffer (ldc=512)
    run_gemm(pool, Wg1, bg1, gg, G, 156, 312, 156, 1, s);
    run_gemm(gg, Wg2, bg2, xc_out, G, 128, 156, 512, 0, s);
}

extern "C" void kernel_launch(void* const* d_in, const int* in_sizes, int n_in,
                              void* d_out, int out_size, void* d_ws, size_t ws_size,
                              hipStream_t stream) {
    const float* x1     = (const float*)d_in[0];
    const int*   ei1    = (const int*)  d_in[1];
    const int*   batch1 = (const int*)  d_in[2];
    const float* x2     = (const float*)d_in[3];
    const int*   ei2    = (const int*)  d_in[4];
    const int*   batch2 = (const int*)  d_in[5];
    const float* cell   = (const float*)d_in[6];
    const float* Wc1 = (const float*)d_in[7];  const float* bc1 = (const float*)d_in[8];
    const float* Wc2 = (const float*)d_in[9];  const float* bc2 = (const float*)d_in[10];
    const float* Wc3 = (const float*)d_in[11]; const float* bc3 = (const float*)d_in[12];
    const float* Wg1 = (const float*)d_in[13]; const float* bg1 = (const float*)d_in[14];
    const float* Wg2 = (const float*)d_in[15]; const float* bg2 = (const float*)d_in[16];
    const float* Wr1 = (const float*)d_in[17]; const float* br1 = (const float*)d_in[18];
    const float* Wr2 = (const float*)d_in[19]; const float* br2 = (const float*)d_in[20];
    const float* Wr3 = (const float*)d_in[21]; const float* br3 = (const float*)d_in[22];
    const float* Wf1 = (const float*)d_in[23]; const float* bf1 = (const float*)d_in[24];
    const float* Wf2 = (const float*)d_in[25]; const float* bf2 = (const float*)d_in[26];
    const float* Wf3 = (const float*)d_in[27]; const float* bf3 = (const float*)d_in[28];
    const float* Wo  = (const float*)d_in[29]; const float* bo  = (const float*)d_in[30];
    float* out = (float*)d_out;

    const int N = in_sizes[0] / 78;   // 20000
    const int E = in_sizes[1] / 2;    // 320000
    const int G = in_sizes[6] / 954;  // 256

    // workspace carve (floats)
    float* p = (float*)d_ws;
    float* t0    = p; p += (size_t)N * 312;
    float* t1    = p; p += (size_t)N * 312;
    float* deg   = p; p += N;
    float* dinv  = p; p += N;
    float* pool1 = p; p += (size_t)G * 312;
    float* pool2 = p; p += (size_t)G * 312;
    float* gg    = p; p += (size_t)G * 156;
    float* xc    = p; p += (size_t)G * 512;
    float* cv0   = p; p += (size_t)G * 2048;
    float* cv1   = p; p += (size_t)G * 512;
    float* cvn   = p; p += (size_t)G * 954;
    float* f1    = p; p += (size_t)G * 1024;
    float* f2    = p; p += (size_t)G * 512;
    float* f3    = p; p += (size_t)G * 128;
    (void)ws_size; (void)n_in; (void)out_size;

    // two drug branches -> xc[:, 0:128] and xc[:, 128:256]
    drug_branch(x1, ei1, batch1, Wc1, bc1, Wc2, bc2, Wc3, bc3, Wg1, bg1, Wg2, bg2,
                t0, t1, deg, dinv, pool1, gg, xc + 0, N, E, G, stream);
    drug_branch(x2, ei2, batch2, Wc1, bc1, Wc2, bc2, Wc3, bc3, Wg1, bg1, Wg2, bg2,
                t0, t1, deg, dinv, pool2, gg, xc + 128, N, E, G, stream);

    // cell line branch -> xc[:, 256:512]
    l2norm_kernel<<<G, 256, 0, stream>>>(cell, cvn, 954);
    run_gemm(cvn, Wr1, br1, cv0, G, 2048, 954, 2048, 1, stream);
    run_gemm(cv0, Wr2, br2, cv1, G, 512, 2048, 512, 1, stream);
    run_gemm(cv1, Wr3, br3, xc + 256, G, 256, 512, 512, 1, stream);

    // head MLP 512 -> 1024 -> 512 -> 128 -> 2
    run_gemm(xc, Wf1, bf1, f1, G, 1024, 512, 1024, 1, stream);
    run_gemm(f1, Wf2, bf2, f2, G, 512, 1024, 512, 1, stream);
    run_gemm(f2, Wf3, bf3, f3, G, 128, 512, 128, 1, stream);
    run_gemm(f3, Wo, bo, out, G, 2, 128, 2, 0, stream);
}
